// TensorProductScoreModel_72387378807427
// MI455X (gfx1250) — matrix operand order repaired and checked by reading, not compiled, not run on hardware
//
#include <hip/hip_runtime.h>

typedef __bf16 bf16_t;
typedef __attribute__((ext_vector_type(16))) __bf16 v16bf;
typedef __attribute__((ext_vector_type(8)))  float  v8f;
typedef __attribute__((ext_vector_type(4))) unsigned int u32x4;
typedef __attribute__((ext_vector_type(8))) int i32x8;
typedef __attribute__((ext_vector_type(4))) int i32x4;

#define HSTRIDE 56

// ---------------- sizes from the reference ----------------
enum { NLIG = 4096, NREC = 16384, ELIG = 32768, EREC = 65536, ECROSS = 32768, NG = 64 };

// ---------------- atomics ----------------
__device__ inline void atomAddF(float* p, float v) {
  __hip_atomic_fetch_add(p, v, __ATOMIC_RELAXED, __HIP_MEMORY_SCOPE_AGENT);
}

// ---------------- TDM: async 2-D f32 tile load global -> LDS ----------------
// D# layout per CDNA5 ISA 8.3/8.4: group0 = {count/type/lds_addr/global_addr},
// group1 = {data_size, tensor_dim0/1, tile_dim0/1, tensor_dim0_stride}. 2-D only,
// groups 2/3 zero. Issued wave-level (EXEC ignored); tracked on TENSORcnt.
// This toolchain exposes the 6-arg builtin: (g0, g1, g2, g3, g4, cpol).
__device__ inline void tdm_load_2d_f32(const float* gsrc, void* lds_dst,
                                       int tile_w, int tile_h,
                                       int tensor_w, int tensor_h, int stride_elems) {
  unsigned long long ga = (unsigned long long)(const void*)gsrc;
  unsigned lds_off = (unsigned)(unsigned long long)lds_dst;
  u32x4 g0;
  g0[0] = 1u;                                            // count = 1 valid descriptor
  g0[1] = lds_off;                                       // LDS byte address
  g0[2] = (unsigned)(ga & 0xffffffffu);                  // global addr lo
  g0[3] = (unsigned)((ga >> 32) & 0x01ffffffu) | (2u << 30); // addr hi | type=2
  i32x8 g1;
  g1[0] = (2 << 16);                                     // data_size = 2 (4 bytes)
  g1[1] = (tensor_w & 0xffff) << 16;                     // tensor_dim0[15:0]
  g1[2] = ((tensor_w >> 16) & 0xffff) | ((tensor_h & 0xffff) << 16);
  g1[3] = ((tensor_h >> 16) & 0xffff) | ((tile_w & 0xffff) << 16);
  g1[4] = (tile_h & 0xffff);                             // tile_dim1 (tile_dim2 = 0)
  g1[5] = stride_elems;                                  // tensor_dim0_stride lo32
  g1[6] = 0;
  g1[7] = 0;
  i32x4 z4 = {0, 0, 0, 0};
  i32x8 z8 = {0, 0, 0, 0, 0, 0, 0, 0};
  __builtin_amdgcn_tensor_load_to_lds(g0, g1, z4, z4, z8, 0);
}
__device__ inline void tdm_wait() { __builtin_amdgcn_s_wait_tensorcnt(0); }

// ---------------- WMMA helpers (wave32, bf16 16x16x32 -> f32) ----------------
// A fragment: 16x32 bf16 tile, row M = lane&15, K per ISA layout table.
__device__ inline v16bf load_a_frag(const float* __restrict__ A, int lda, int lane, int kbase) {
  v16bf a;
  const int m = lane & 15;
  const int kofs = (lane >> 4) << 3;
#pragma unroll
  for (int e = 0; e < 16; ++e) {
    int k = kbase + ((e & 8) << 1) + (e & 7) + kofs;
    a[e] = (bf16_t)A[m * lda + k];
  }
  return a;
}
// B fragment: 32x16 bf16 tile, col N = lane&15, K per ISA layout table (same as A).
__device__ inline v16bf load_b_frag(const float* __restrict__ B, int ldb, int lane, int kbase) {
  v16bf b;
  const int n = lane & 15;
  const int kofs = (lane >> 4) << 3;
#pragma unroll
  for (int e = 0; e < 16; ++e) {
    int k = kbase + ((e & 8) << 1) + (e & 7) + kofs;
    b[e] = (bf16_t)B[k * ldb + n];
  }
  return b;
}
__device__ inline v8f wmma_bf16(v16bf a, v16bf b, v8f c) {
  return __builtin_amdgcn_wmma_f32_16x16x32_bf16(false, a, false, b, (short)0, c, false, false);
}
__device__ inline v8f v8f_zero() {
  v8f z = {0.f, 0.f, 0.f, 0.f, 0.f, 0.f, 0.f, 0.f};
  return z;
}

// ---------------- tensor-product path tables ----------------
struct PathD {
  short x_off, m1, n1;      // input irrep slice: m1 channels of width n1=2l1+1
  short sh_off, n2;         // SH slice
  short z_off, m3, n3;      // output irrep slice
  short cg;                 // offset into CG[]
  short w_off;              // weight offset within wn
  float norm;               // 1/sqrt(fan_in)
};

// CG blocks: 0:(000)[1] 1:(011)[9] 10:(101)[9] 19:(110)[9] 28:(111)[27] 55:(121)[45]
__constant__ float CG[100] = {
  // (0,0,0)
  1.f,
  // (0,1,1) identity
  1.f,0.f,0.f, 0.f,1.f,0.f, 0.f,0.f,1.f,
  // (1,0,1) identity
  1.f,0.f,0.f, 0.f,1.f,0.f, 0.f,0.f,1.f,
  // (1,1,0) delta/sqrt(3)
  0.5773503f,0.f,0.f, 0.f,0.5773503f,0.f, 0.f,0.f,0.5773503f,
  // (1,1,1) eps/sqrt(2): flat (i*3+j)*3+k
  0.f,0.f,0.f, 0.f,0.f,0.7071068f, 0.f,-0.7071068f,0.f,
  0.f,0.f,-0.7071068f, 0.f,0.f,0.f, 0.7071068f,0.f,0.f,
  0.f,0.7071068f,0.f, -0.7071068f,0.f,0.f, 0.f,0.f,0.f,
  // (1,2,1): 3x5x3 = 45 entries, flat (i*5+j)*3+k  (Q.r projection structure)
  0.f,0.f,0.5477226f,   0.f,0.5477226f,0.f,   -0.3162278f,0.f,0.f,
  0.f,0.f,0.f,          -0.5477226f,0.f,0.f,  0.f,0.f,0.f,
  0.5477226f,0.f,0.f,   0.f,0.6324555f,0.f,   0.f,0.f,0.5477226f,
  0.f,0.f,0.f,          0.5477226f,0.f,0.f,   0.f,0.f,0.f,
  0.f,0.f,-0.3162278f,  0.f,0.5477226f,0.f,   0.f,0.f,0.5477226f
};

__constant__ PathD PATHS[18] = {
  // layer 0 (2 paths), fan: 16,16
  {0,16,1, 0,1, 0,16,1, 0,   0, 0.25f},
  {0,16,1, 1,3, 16,4,3, 1, 256, 0.25f},
  // layer 1 (6 paths), fan: 20,24,4
  {0,16,1, 0,1, 0,16,1,  0,   0, 0.2236068f},
  {0,16,1, 1,3, 16,4,3,  1, 256, 0.2041241f},
  {16,4,3, 0,1, 16,4,3, 10, 320, 0.2041241f},
  {16,4,3, 1,3, 0,16,1, 19, 336, 0.2236068f},
  {16,4,3, 1,3, 28,4,3, 28, 400, 0.5f},
  {16,4,3, 4,5, 16,4,3, 55, 416, 0.2041241f},
  // layer 2 (10 paths), fan: 20,28,12,4
  {0,16,1, 0,1, 0,16,1,  0,   0, 0.2236068f},
  {0,16,1, 1,3, 16,4,3,  1, 256, 0.1889822f},
  {16,4,3, 0,1, 16,4,3, 10, 320, 0.1889822f},
  {16,4,3, 1,3, 0,16,1, 19, 336, 0.2236068f},
  {16,4,3, 1,3, 28,4,3, 28, 400, 0.2886751f},
  {16,4,3, 4,5, 16,4,3, 55, 416, 0.1889822f},
  {28,4,3, 0,1, 28,4,3, 10, 432, 0.2886751f},
  {28,4,3, 1,3, 16,4,3, 28, 448, 0.1889822f},
  {28,4,3, 1,3, 40,16,1,19, 464, 0.5f},
  {28,4,3, 4,5, 28,4,3, 55, 528, 0.2886751f}
};
__constant__ int PATH_START[3] = {0, 2, 8};
__constant__ int PATH_COUNT[3] = {2, 6, 10};

// ================= kernel: mlp2 on edges (din -> 16 relu -> 16) ===============
// block = 128 threads (4 waves), 64 rows per block, K padded to K1 (<=96)
__global__ void edge_mlp2_kernel(const float* __restrict__ X, int din, int K1,
                                 const float* __restrict__ W1, const float* __restrict__ b1,
                                 const float* __restrict__ W2, const float* __restrict__ b2,
                                 float* __restrict__ Y, int E) {
  __shared__ float sX[64 * 96];
  __shared__ float sW[96 * 16];
  __shared__ float sH[64 * 32];
  __shared__ float sW2[32 * 16];
  const int eb = blockIdx.x * 64;
  const int tid = threadIdx.x;
  for (int idx = tid; idx < 64 * K1; idx += 128) {
    int r = idx / K1, c = idx % K1;
    sX[r * K1 + c] = (c < din) ? X[(size_t)(eb + r) * din + c] : 0.f;
  }
  for (int idx = tid; idx < K1 * 16; idx += 128) {
    int k = idx >> 4, n = idx & 15;
    sW[idx] = (k < din) ? W1[k * 16 + n] : 0.f;
  }
  for (int idx = tid; idx < 64 * 32; idx += 128) sH[idx] = 0.f;
  for (int idx = tid; idx < 32 * 16; idx += 128) {
    int k = idx >> 4, n = idx & 15;
    sW2[idx] = (k < 16) ? W2[k * 16 + n] : 0.f;
  }
  __syncthreads();
  const int wave = tid >> 5, lane = tid & 31;
  const int n = lane & 15, mo = (lane >> 4) << 3;
  // GEMM1: [16 x K1] x [K1 x 16]
  {
    v8f acc = v8f_zero();
    const float* Arow = sX + wave * 16 * K1;
    for (int kb = 0; kb < K1; kb += 32) {
      v16bf a = load_a_frag(Arow, K1, lane, kb);
      v16bf b = load_b_frag(sW, 16, lane, kb);
      acc = wmma_bf16(a, b, acc);
    }
    float bias = b1[n];
#pragma unroll
    for (int r = 0; r < 8; ++r)
      sH[(wave * 16 + mo + r) * 32 + n] = fmaxf(acc[r] + bias, 0.f);
  }
  __syncthreads();
  // GEMM2: [16 x 32(pad)] x [32 x 16]
  {
    v8f acc = v8f_zero();
    const float* Arow = sH + wave * 16 * 32;
    v16bf a = load_a_frag(Arow, 32, lane, 0);
    v16bf b = load_b_frag(sW2, 16, lane, 0);
    acc = wmma_bf16(a, b, acc);
    float bias = b2[n];
#pragma unroll
    for (int r = 0; r < 8; ++r) {
      int row = eb + wave * 16 + mo + r;
      if (row < E) Y[(size_t)row * 16 + n] = acc[r] + bias;
    }
  }
}

// ================= kernel: spherical harmonics (l<=2) ===============
__global__ void sph_kernel(const float* __restrict__ vec, float* __restrict__ sh, int E) {
  int e = blockIdx.x * blockDim.x + threadIdx.x;
  if (e >= E) return;
  float x = vec[e * 3 + 0], y = vec[e * 3 + 1], z = vec[e * 3 + 2];
  float r = sqrtf(x * x + y * y + z * z) + 1e-9f;
  x /= r; y /= r; z /= r;
  const float s3 = 1.7320508f, s15 = 3.8729833f, s5 = 2.2360680f;
  float* o = sh + (size_t)e * 9;
  o[0] = 1.f;       o[1] = s3 * y;    o[2] = s3 * z;    o[3] = s3 * x;
  o[4] = s15 * x * y; o[5] = s15 * y * z; o[6] = 0.5f * s5 * (3.f * z * z - 1.f);
  o[7] = s15 * x * z; o[8] = 0.5f * s15 * (x * x - y * y);
}

// ================= kernel: fused tp_conv ===============
// block = 128 threads (4 wave32), 64 edges / block.
// stages: gather e -> h1 -> h2 -> w (three WMMA GEMM stages; mlp weights staged
// via TDM tensor_load_to_lds, waited on TENSORcnt), then per-edge CG tensor
// product (VALU) + atomic scatter-add.
__global__ void tp_conv_kernel(const float* __restrict__ hs, const float* __restrict__ hd,
                               const int* __restrict__ src, const int* __restrict__ dst,
                               const float* __restrict__ esh, const float* __restrict__ ea,
                               const float* __restrict__ W1, const float* __restrict__ b1,
                               const float* __restrict__ W2, const float* __restrict__ b2,
                               const float* __restrict__ W3, const float* __restrict__ b3,
                               float* __restrict__ out_sum,
                               int E, int din, int wn, int layer) {
  extern __shared__ float smem[];
  float* sE   = smem;               // 64 x 64 (K pad 48->64; cols 48..63 stay 0)
  float* sH   = sE + 64 * 64;       // 64 x 64
  float* sWm  = sH + 64 * 64;       // 64 x 48 (K-major, rows 48..63 zero)
  float* sW3t = sWm + 64 * 48;      // 64 x 16 (one N-tile of W3, rows 48..63 zero)
  float* sXg  = sW3t + 64 * 16;     // 64 x 40 (gathered source features)
  float* sSh  = sXg + 64 * 40;      // 64 x 9
  bf16_t* sWout = (bf16_t*)(sSh + 64 * 9); // 64 x wn (<=544) bf16 per-edge weights

  const int eb = blockIdx.x * 64;
  const int tid = threadIdx.x;
  const int wave = tid >> 5, lane = tid & 31;
  const int n = lane & 15, mo = (lane >> 4) << 3;

  // ---- async: W1 (48x48 f32) -> sWm rows 0..47 via Tensor Data Mover ----
  if (tid == 0) tdm_load_2d_f32(W1, sWm, 48, 48, 48, 48, 48);

  // ---- stage edge embedding [ea | hs[src,:16] | hd[dst,:16]] and misc ----
  for (int idx = tid; idx < 64 * 64; idx += 128) {
    int r = idx >> 6, c = idx & 63;
    int e = eb + r;
    float v = 0.f;
    if (c < 16)      v = ea[(size_t)e * 16 + c];
    else if (c < 32) v = hs[(size_t)src[e] * HSTRIDE + (c - 16)];
    else if (c < 48) v = hd[(size_t)dst[e] * HSTRIDE + (c - 32)];
    sE[idx] = v;
    sH[idx] = 0.f;
  }
  for (int idx = tid; idx < 64 * 40; idx += 128) {
    int r = idx / 40, c = idx % 40;
    sXg[idx] = (c < din) ? hs[(size_t)src[eb + r] * HSTRIDE + c] : 0.f;
  }
  for (int idx = tid; idx < 64 * 9; idx += 128)
    sSh[idx] = esh[(size_t)(eb + idx / 9) * 9 + idx % 9];
  // zero K-padding rows 48..63 of the weight tiles (TDM fills rows 0..47)
  for (int idx = tid; idx < 16 * 48; idx += 128) sWm[48 * 48 + idx] = 0.f;
  for (int idx = tid; idx < 16 * 16; idx += 128) sW3t[48 * 16 + idx] = 0.f;
  tdm_wait();
  __syncthreads();

  // ---- GEMM1: h1 = relu(e @ W1 + b1), [64x64(pad)] x [64x48] ----
  {
    const float* Arow = sE + wave * 16 * 64;
#pragma unroll
    for (int nt = 0; nt < 3; ++nt) {
      v8f acc = v8f_zero();
#pragma unroll
      for (int kb = 0; kb < 64; kb += 32) {
        v16bf a = load_a_frag(Arow, 64, lane, kb);
        v16bf b = load_b_frag(sWm + nt * 16, 48, lane, kb);
        acc = wmma_bf16(a, b, acc);
      }
      float bias = b1[nt * 16 + n];
#pragma unroll
      for (int r = 0; r < 8; ++r)
        sH[(wave * 16 + mo + r) * 64 + nt * 16 + n] = fmaxf(acc[r] + bias, 0.f);
    }
  }
  __syncthreads();
  // ---- async: W2 -> sWm (overwrites rows 0..47) ----
  if (tid == 0) tdm_load_2d_f32(W2, sWm, 48, 48, 48, 48, 48);
  tdm_wait();
  __syncthreads();
  // ---- GEMM2: h2 = relu(h1 @ W2 + b2) -> overwrite sE cols 0..47 ----
  {
    const float* Arow = sH + wave * 16 * 64;
#pragma unroll
    for (int nt = 0; nt < 3; ++nt) {
      v8f acc = v8f_zero();
#pragma unroll
      for (int kb = 0; kb < 64; kb += 32) {
        v16bf a = load_a_frag(Arow, 64, lane, kb);
        v16bf b = load_b_frag(sWm + nt * 16, 48, lane, kb);
        acc = wmma_bf16(a, b, acc);
      }
      float bias = b2[nt * 16 + n];
#pragma unroll
      for (int r = 0; r < 8; ++r)
        sE[(wave * 16 + mo + r) * 64 + nt * 16 + n] = fmaxf(acc[r] + bias, 0.f);
    }
  }
  __syncthreads();
  // ---- GEMM3: w = h2 @ W3 + b3, N tiled by 16; W3 tiles streamed via TDM ----
  const int ntiles = wn >> 4;
  for (int nt = 0; nt < ntiles; ++nt) {
    if (tid == 0) tdm_load_2d_f32(W3 + nt * 16, sW3t, 16, 48, wn, 48, wn);
    tdm_wait();
    __syncthreads();
    v8f acc = v8f_zero();
    const float* Arow = sE + wave * 16 * 64;
#pragma unroll
    for (int kb = 0; kb < 64; kb += 32) {
      v16bf a = load_a_frag(Arow, 64, lane, kb);
      v16bf b = load_b_frag(sW3t, 16, lane, kb);
      acc = wmma_bf16(a, b, acc);
    }
    float bias = b3[nt * 16 + n];
#pragma unroll
    for (int r = 0; r < 8; ++r)
      sWout[(size_t)(wave * 16 + mo + r) * wn + nt * 16 + n] = (bf16_t)(acc[r] + bias);
    __syncthreads();
  }

  // ---- per-edge CG tensor product (VALU) + scatter ----
  // 2 threads per edge: half 0 handles out channels [0,28), half 1 [28,56).
  const int er = tid >> 1;
  const int half = tid & 1;
  float z[28];
#pragma unroll
  for (int c = 0; c < 28; ++c) z[c] = 0.f;
  const int ps = PATH_START[layer], pc = PATH_COUNT[layer];
  for (int p = ps; p < ps + pc; ++p) {
    PathD pd = PATHS[p];
    if (half == 0) { if (pd.z_off >= 28) continue; }
    else           { if (pd.z_off <  28) continue; }
    const int zbase = pd.z_off - half * 28;
    const float* cg = CG + pd.cg;
    float y[5];
    for (int j = 0; j < pd.n2; ++j) y[j] = sSh[er * 9 + pd.sh_off + j];
    for (int u = 0; u < pd.m1; ++u) {
      float xv[3];
      for (int i = 0; i < pd.n1; ++i) xv[i] = sXg[er * 40 + pd.x_off + u * pd.n1 + i];
      for (int v = 0; v < pd.m3; ++v) {
        float wv = (float)sWout[(size_t)er * wn + pd.w_off + u * pd.m3 + v] * pd.norm;
        for (int i = 0; i < pd.n1; ++i)
          for (int j = 0; j < pd.n2; ++j) {
            float xyw = xv[i] * y[j] * wv;
            const float* crow = cg + (i * pd.n2 + j) * pd.n3;
            for (int k = 0; k < pd.n3; ++k)
              z[zbase + v * pd.n3 + k] += crow[k] * xyw;
          }
      }
    }
  }
  if (eb + er < E) {
    int d = dst[eb + er];
    float* o = out_sum + (size_t)d * HSTRIDE + half * 28;
    for (int c = 0; c < 28; ++c) atomAddF(&o[c], z[c]);
  }
}

// ================= small kernels ===============
__global__ void count_kernel(const int* __restrict__ dst, int E, float* __restrict__ cnt) {
  int e = blockIdx.x * blockDim.x + threadIdx.x;
  if (e < E) atomAddF(&cnt[dst[e]], 1.f);
}
__global__ void init_h_kernel(const float* __restrict__ x, float* __restrict__ h, int N) {
  int i = blockIdx.x * blockDim.x + threadIdx.x;
  if (i >= N * HSTRIDE) return;
  int nn = i / HSTRIDE, c = i % HSTRIDE;
  h[i] = (c < 16) ? x[nn * 16 + c] : 0.f;
}
__global__ void finalize_kernel(const float* __restrict__ h_old, int prev_dim,
                                const float* __restrict__ accA, const float* __restrict__ cntA,
                                const float* __restrict__ accB, const float* __restrict__ cntB,
                                float* __restrict__ h_new, int N) {
  int i = blockIdx.x * blockDim.x + threadIdx.x;
  if (i >= N * HSTRIDE) return;
  int nn = i / HSTRIDE, c = i % HSTRIDE;
  float v = (c < prev_dim) ? h_old[i] : 0.f;
  v += accA[i] / fmaxf(cntA[nn], 1.f);
  v += accB[i] / fmaxf(cntB[nn], 1.f);
  h_new[i] = v;
}
__global__ void pool_kernel(const float* __restrict__ h, const int* __restrict__ batch,
                            float* __restrict__ pooled, float* __restrict__ gcnt, int N) {
  int nn = blockIdx.x * blockDim.x + threadIdx.x;
  if (nn >= N) return;
  int g = batch[nn];
  for (int c = 0; c < 16; ++c) atomAddF(&pooled[g * 32 + c], h[(size_t)nn * HSTRIDE + c]);
  for (int c = 0; c < 16; ++c) atomAddF(&pooled[g * 32 + 16 + c], h[(size_t)nn * HSTRIDE + 40 + c]);
  atomAddF(&gcnt[g], 1.f);
}
__global__ void conf_kernel(const float* __restrict__ pooled, const float* __restrict__ gcnt,
                            const float* __restrict__ W1, const float* __restrict__ b1,
                            const float* __restrict__ W2, const float* __restrict__ b2,
                            const float* __restrict__ W3, const float* __restrict__ b3,
                            const float* __restrict__ g1, const float* __restrict__ be1,
                            const float* __restrict__ g2, const float* __restrict__ be2,
                            float* __restrict__ out, int G) {
  int g = blockIdx.x * blockDim.x + threadIdx.x;
  if (g >= G) return;
  const float inv = 0.9999950f; // 1/sqrt(1+1e-5)
  float cnt = fmaxf(gcnt[g], 1.f);
  float p[32];
  for (int i = 0; i < 32; ++i) p[i] = pooled[g * 32 + i] / cnt;
  float h1[16];
  for (int o = 0; o < 16; ++o) {
    float s = b1[o];
    for (int i = 0; i < 32; ++i) s += p[i] * W1[i * 16 + o];
    h1[o] = fmaxf(s * inv * g1[o] + be1[o], 0.f);
  }
  float h2[16];
  for (int o = 0; o < 16; ++o) {
    float s = b2[o];
    for (int i = 0; i < 16; ++i) s += h1[i] * W2[i * 16 + o];
    h2[o] = fmaxf(s * inv * g2[o] + be2[o], 0.f);
  }
  float s = b3[0];
  for (int i = 0; i < 16; ++i) s += h2[i] * W3[i];
  out[g] = s;
}

// ================= host ===============
extern "C" void kernel_launch(void* const* d_in, const int* in_sizes, int n_in,
                              void* d_out, int out_size, void* d_ws, size_t ws_size,
                              hipStream_t stream) {
  (void)in_sizes; (void)n_in; (void)out_size; (void)ws_size;
  // --- inputs (insertion order of setup_inputs dict, nested dicts in insertion order) ---
  const float* lig_x     = (const float*)d_in[0];
  const float* lig_eraw  = (const float*)d_in[2];
  const float* rec_eraw  = (const float*)d_in[3];
  const float* cross_eraw= (const float*)d_in[4];
  const float* lig_evec  = (const float*)d_in[5];
  const float* rec_evec  = (const float*)d_in[6];
  const float* cross_evec= (const float*)d_in[7];
  const int*   lig_ei    = (const int*)d_in[8];
  const int*   rec_ei    = (const int*)d_in[9];
  const int*   cross_lig = (const int*)d_in[10];
  const int*   cross_rec = (const int*)d_in[11];
  const int*   lig_batch = (const int*)d_in[12];
  // edge_mlps: lig, rec, cross; each {W1,b1,W2,b2} -> indices 14..25
#define EM(t, w) ((const float*)d_in[14 + (t)*4 + (w)])
  // conv_params: lig(0), rec(1), l2r(2), r2l(3); 3 layers x {W1,b1,W2,b2,W3,b3} -> 26..97
#define CV(t, l, w) ((const float*)d_in[26 + (t)*18 + (l)*6 + (w)])
  // conf_params: 98..107 (W1,b1,W2,b2,W3,b3,g1,be1,g2,be2)
#define CF(w) ((const float*)d_in[98 + (w)])

  float* ws = (float*)d_ws;
  size_t o = 0;
  float* lig_ea   = ws + o; o += (size_t)ELIG * 16;
  float* rec_ea   = ws + o; o += (size_t)EREC * 16;
  float* cross_ea = ws + o; o += (size_t)ECROSS * 16;
  float* lig_sh   = ws + o; o += (size_t)ELIG * 9;
  float* rec_sh   = ws + o; o += (size_t)EREC * 9;
  float* cross_sh = ws + o; o += (size_t)ECROSS * 9;
  float* hligA    = ws + o; o += (size_t)NLIG * HSTRIDE;
  float* hligB    = ws + o; o += (size_t)NLIG * HSTRIDE;
  float* hrecA    = ws + o; o += (size_t)NREC * HSTRIDE;
  float* hrecB    = ws + o; o += (size_t)NREC * HSTRIDE;
  float* acc_base = ws + o;
  float* acc_li   = ws + o; o += (size_t)NLIG * HSTRIDE;
  float* acc_lx   = ws + o; o += (size_t)NLIG * HSTRIDE;
  float* acc_ri   = ws + o; o += (size_t)NREC * HSTRIDE;
  float* acc_rx   = ws + o; o += (size_t)NREC * HSTRIDE;
  size_t acc_bytes = (size_t)(2 * NLIG + 2 * NREC) * HSTRIDE * sizeof(float);
  float* cnt_base = ws + o;
  float* cnt_li   = ws + o; o += NLIG;
  float* cnt_lx   = ws + o; o += NLIG;
  float* cnt_ri   = ws + o; o += NREC;
  float* cnt_rx   = ws + o; o += NREC;
  size_t cnt_bytes = (size_t)(2 * NLIG + 2 * NREC) * sizeof(float);
  float* pooled   = ws + o; o += (size_t)NG * 32;
  float* gcnt     = ws + o; o += NG;

  const int* lig_src = lig_ei, *lig_dst = lig_ei + ELIG;
  const int* rec_src = rec_ei, *rec_dst = rec_ei + EREC;

  // ---- edge attribute MLPs (WMMA) ----
  edge_mlp2_kernel<<<ELIG / 64, 128, 0, stream>>>(lig_eraw, 68, 96, EM(0,0), EM(0,1), EM(0,2), EM(0,3), lig_ea, ELIG);
  edge_mlp2_kernel<<<EREC / 64, 128, 0, stream>>>(rec_eraw, 64, 64, EM(1,0), EM(1,1), EM(1,2), EM(1,3), rec_ea, EREC);
  edge_mlp2_kernel<<<ECROSS / 64, 128, 0, stream>>>(cross_eraw, 64, 64, EM(2,0), EM(2,1), EM(2,2), EM(2,3), cross_ea, ECROSS);

  // ---- spherical harmonics ----
  sph_kernel<<<(ELIG + 255) / 256, 256, 0, stream>>>(lig_evec, lig_sh, ELIG);
  sph_kernel<<<(EREC + 255) / 256, 256, 0, stream>>>(rec_evec, rec_sh, EREC);
  sph_kernel<<<(ECROSS + 255) / 256, 256, 0, stream>>>(cross_evec, cross_sh, ECROSS);

  // ---- node feature init (stride 56, zero padded) ----
  init_h_kernel<<<(NLIG * HSTRIDE + 255) / 256, 256, 0, stream>>>(lig_x, hligA, NLIG);
  init_h_kernel<<<(NREC * HSTRIDE + 255) / 256, 256, 0, stream>>>((const float*)d_in[1], hrecA, NREC);

  // ---- segment-mean denominators (once) ----
  (void)hipMemsetAsync(cnt_base, 0, cnt_bytes, stream);
  count_kernel<<<(ELIG + 255) / 256, 256, 0, stream>>>(lig_dst, ELIG, cnt_li);
  count_kernel<<<(ECROSS + 255) / 256, 256, 0, stream>>>(cross_lig, ECROSS, cnt_lx);
  count_kernel<<<(EREC + 255) / 256, 256, 0, stream>>>(rec_dst, EREC, cnt_ri);
  count_kernel<<<(ECROSS + 255) / 256, 256, 0, stream>>>(cross_rec, ECROSS, cnt_rx);

  // ---- layers ----
  const int din_a[3] = {16, 28, 40};
  const int wn_a[3] = {320, 432, 544};
  const size_t smemTP = (size_t)(64 * 64 * 2 + 64 * 48 + 64 * 16 + 64 * 40 + 64 * 9) * 4 + (size_t)64 * 544 * 2;
  const float* hlig = hligA; float* hlig_n = hligB;
  const float* hrec = hrecA; float* hrec_n = hrecB;
  for (int L = 0; L < 3; ++L) {
    int din = din_a[L], wn = wn_a[L];
    (void)hipMemsetAsync(acc_base, 0, acc_bytes, stream);
    // lig intra
    tp_conv_kernel<<<ELIG / 64, 128, smemTP, stream>>>(hlig, hlig, lig_src, lig_dst, lig_sh, lig_ea,
        CV(0,L,0), CV(0,L,1), CV(0,L,2), CV(0,L,3), CV(0,L,4), CV(0,L,5), acc_li, ELIG, din, wn, L);
    // r2l: src = cross_rec (rec feats), dst = cross_lig
    tp_conv_kernel<<<ECROSS / 64, 128, smemTP, stream>>>(hrec, hlig, cross_rec, cross_lig, cross_sh, cross_ea,
        CV(3,L,0), CV(3,L,1), CV(3,L,2), CV(3,L,3), CV(3,L,4), CV(3,L,5), acc_lx, ECROSS, din, wn, L);
    if (L < 2) {
      // rec intra
      tp_conv_kernel<<<EREC / 64, 128, smemTP, stream>>>(hrec, hrec, rec_src, rec_dst, rec_sh, rec_ea,
          CV(1,L,0), CV(1,L,1), CV(1,L,2), CV(1,L,3), CV(1,L,4), CV(1,L,5), acc_ri, EREC, din, wn, L);
      // l2r: src = cross_lig (lig feats), dst = cross_rec
      tp_conv_kernel<<<ECROSS / 64, 128, smemTP, stream>>>(hlig, hrec, cross_lig, cross_rec, cross_sh, cross_ea,
          CV(2,L,0), CV(2,L,1), CV(2,L,2), CV(2,L,3), CV(2,L,4), CV(2,L,5), acc_rx, ECROSS, din, wn, L);
    }
    finalize_kernel<<<(NLIG * HSTRIDE + 255) / 256, 256, 0, stream>>>(hlig, din, acc_li, cnt_li, acc_lx, cnt_lx, hlig_n, NLIG);
    { const float* t = hlig; hlig = hlig_n; hlig_n = (float*)t; }
    if (L < 2) {
      finalize_kernel<<<(NREC * HSTRIDE + 255) / 256, 256, 0, stream>>>(hrec, din, acc_ri, cnt_ri, acc_rx, cnt_rx, hrec_n, NREC);
      const float* t = hrec; hrec = hrec_n; hrec_n = (float*)t;
    }
  }

  // ---- pooling + confidence MLP ----
  (void)hipMemsetAsync(pooled, 0, (size_t)(NG * 32 + NG) * sizeof(float), stream);
  pool_kernel<<<(NLIG + 255) / 256, 256, 0, stream>>>(hlig, lig_batch, pooled, gcnt, NLIG);
  conf_kernel<<<1, 64, 0, stream>>>(pooled, gcnt, CF(0), CF(1), CF(2), CF(3), CF(4), CF(5),
                                    CF(6), CF(7), CF(8), CF(9), (float*)d_out, NG);
}
